// GAT_47029891891201
// MI455X (gfx1250) — matrix operand order repaired and checked by reading, not compiled, not run on hardware
//
#include <hip/hip_runtime.h>
#include <hip/hip_bf16.h>
#include <math.h>

#define N_NODES  50000
#define N_EDGES  1600000
#define NFEAT    256
#define HIDALL   256     // HEADS*NHID
#define HEADS    8
#define NHID     32
#define NCLASS   2
#define NEG_SLOPE 0.2f

typedef __attribute__((ext_vector_type(16))) __bf16 v16bf;
typedef __attribute__((ext_vector_type(8)))  float  v8f;

union FragBF { v16bf v; uint4 q[2]; };

__device__ __forceinline__ unsigned short f2bf(float f) {
  unsigned int u = __float_as_uint(f);
  u += 0x7fffu + ((u >> 16) & 1u);        // round-to-nearest-even
  return (unsigned short)(u >> 16);
}

// monotonic-int atomic float max (init with -inf)
__device__ __forceinline__ void atomicMaxF(float* addr, float v) {
  if (v >= 0.0f) atomicMax((int*)addr, __float_as_int(v));
  else           atomicMin((unsigned int*)addr, __float_as_uint(v));
}

// ---------------- utility kernels ----------------
__global__ void k_fill(float* __restrict__ p, float v, int n) {
  int i = blockIdx.x * 256 + threadIdx.x;
  if (i < n) p[i] = v;
}

__global__ void k_f32_to_bf16(const float* __restrict__ in,
                              unsigned short* __restrict__ out, int n) {
  int i = blockIdx.x * 256 + threadIdx.x;
  if (i < n) out[i] = f2bf(in[i]);
}

// ---------------- layer-1 GEMM: feat = x @ W1 (bf16 WMMA, f32 acc) ------
__global__ void __launch_bounds__(256)
k_gemm_bf16_wmma(const unsigned short* __restrict__ A,   // [M,K] bf16
                 const unsigned short* __restrict__ B,   // [K,N] bf16
                 float* __restrict__ C, int M, int K, int N) {
  const int lane = threadIdx.x & 31;
  const int wave = threadIdx.x >> 5;
  const int m0 = blockIdx.x * 64 + (wave & 3) * 16;   // 4 waves in M
  const int n0 = blockIdx.y * 64 + (wave >> 2) * 32;  // 2 waves in N, 32 cols each
  int arow = m0 + (lane & 15);
  if (arow >= M) arow = M - 1;                        // clamp, store masked below
  const int akoff = (lane >> 4) << 3;                 // lanes 16-31: K+8
  v8f c0 = {0.f,0.f,0.f,0.f,0.f,0.f,0.f,0.f};
  v8f c1 = {0.f,0.f,0.f,0.f,0.f,0.f,0.f,0.f};
  for (int kb = 0; kb < K; kb += 32) {
    FragBF a, b0, b1;
    const unsigned short* ap = A + (size_t)arow * K + kb + akoff;
    a.q[0] = *(const uint4*)(ap);        // K = kb+akoff   .. +7
    a.q[1] = *(const uint4*)(ap + 16);   // K = kb+akoff+16.. +23
    const unsigned short* bp = B + (size_t)(kb + akoff) * N + n0 + (lane & 15); // lane = N col
    unsigned short* pb0 = (unsigned short*)&b0;
    unsigned short* pb1 = (unsigned short*)&b1;
    #pragma unroll
    for (int s = 0; s < 16; ++s) {
      const int kk = (s < 8) ? s : s + 8;
      pb0[s] = bp[(size_t)kk * N];
      pb1[s] = bp[(size_t)kk * N + 16];
    }
    c0 = __builtin_amdgcn_wmma_f32_16x16x32_bf16(false, a.v, false, b0.v,
                                                 (short)0, c0, false, false);
    c1 = __builtin_amdgcn_wmma_f32_16x16x32_bf16(false, a.v, false, b1.v,
                                                 (short)0, c1, false, false);
  }
  const int rbase = m0 + ((lane >> 4) << 3);
  const int col   = lane & 15;
  #pragma unroll
  for (int v = 0; v < 8; ++v) {
    int r = rbase + v;
    if (r < M) {
      C[(size_t)r * N + n0 + col]      = c0[v];
      C[(size_t)r * N + n0 + 16 + col] = c1[v];
    }
  }
}

// ---------------- el/er: per (node, head) dot over 32 feats -------------
__global__ void k_elr(const float* __restrict__ feat, const float* __restrict__ al,
                      const float* __restrict__ ar, float* __restrict__ el,
                      float* __restrict__ er, int N) {
  int idx = blockIdx.x * 256 + threadIdx.x;       // n*8 + h
  if (idx >= N * HEADS) return;
  int n = idx >> 3, h = idx & 7;
  const float4* fp = (const float4*)(feat + (size_t)n * HIDALL + h * NHID);
  const float4* ap = (const float4*)(al + h * NHID);
  const float4* bp = (const float4*)(ar + h * NHID);
  float sl = 0.f, sr = 0.f;
  #pragma unroll
  for (int t = 0; t < 8; ++t) {
    float4 f = fp[t], a = ap[t], b = bp[t];
    sl += f.x*a.x + f.y*a.y + f.z*a.z + f.w*a.w;
    sr += f.x*b.x + f.y*b.y + f.z*b.z + f.w*b.w;
  }
  el[idx] = sl; er[idx] = sr;
}

// ---------------- layer-1 edge phase 1: score + segment max -------------
__global__ void k_edge1_score(const int* __restrict__ src, const int* __restrict__ dst,
                              const float* __restrict__ el, const float* __restrict__ er,
                              float* __restrict__ e1, float* __restrict__ m1, int E) {
  int e = blockIdx.x * 256 + threadIdx.x;
  if (e >= E) return;
  int s = src[e], d = dst[e];
  const float4* lp = (const float4*)(el + (size_t)s * 8);
  const float4* rp = (const float4*)(er + (size_t)d * 8);
  float4 l0 = lp[0], l1 = lp[1], r0 = rp[0], r1 = rp[1];
  float v[8] = { l0.x+r0.x, l0.y+r0.y, l0.z+r0.z, l0.w+r0.w,
                 l1.x+r1.x, l1.y+r1.y, l1.z+r1.z, l1.w+r1.w };
  #pragma unroll
  for (int h = 0; h < 8; ++h) {
    float x = v[h]; x = x > 0.f ? x : NEG_SLOPE * x; v[h] = x;
    atomicMaxF(m1 + (size_t)d * 8 + h, x);
  }
  float4* ep = (float4*)(e1 + (size_t)e * 8);
  ep[0] = make_float4(v[0], v[1], v[2], v[3]);
  ep[1] = make_float4(v[4], v[5], v[6], v[7]);
}

// ---------------- layer-1 edge phase 2: exp + segment sum ---------------
__global__ void k_edge1_expsum(const int* __restrict__ dst, float* __restrict__ e1,
                               const float* __restrict__ m1, float* __restrict__ d1, int E) {
  int e = blockIdx.x * 256 + threadIdx.x;
  if (e >= E) return;
  int d = dst[e];
  float4* ep = (float4*)(e1 + (size_t)e * 8);
  const float4* mp = (const float4*)(m1 + (size_t)d * 8);
  float4 e0 = ep[0], e1v = ep[1], m0 = mp[0], m1v = mp[1];
  float v[8] = { expf(e0.x-m0.x), expf(e0.y-m0.y), expf(e0.z-m0.z), expf(e0.w-m0.w),
                 expf(e1v.x-m1v.x), expf(e1v.y-m1v.y), expf(e1v.z-m1v.z), expf(e1v.w-m1v.w) };
  #pragma unroll
  for (int h = 0; h < 8; ++h) atomicAdd(d1 + (size_t)d * 8 + h, v[h]);
  ep[0] = make_float4(v[0], v[1], v[2], v[3]);
  ep[1] = make_float4(v[4], v[5], v[6], v[7]);
}

// ---------------- layer-1 edge phase 3: weighted scatter (wave/edge) ----
__global__ void __launch_bounds__(256)
k_edge1_aggregate(const int* __restrict__ src, const int* __restrict__ dst,
                  const float* __restrict__ feat, const float* __restrict__ ee,
                  const float* __restrict__ d1, float* __restrict__ out1, int E) {
  int e = blockIdx.x * 8 + (threadIdx.x >> 5);
  if (e >= E) return;
  int lane = threadIdx.x & 31;
  int s = src[e], d = dst[e];
  int h = lane >> 2;                                   // 8 feats/lane stay in one head
  float alpha = ee[(size_t)e * 8 + h] / fmaxf(d1[(size_t)d * 8 + h], 1e-16f);
  const float4* fp = (const float4*)(feat + (size_t)s * HIDALL + lane * 8);
  float4 f0 = fp[0], f1 = fp[1];
  float* op = out1 + (size_t)d * HIDALL + lane * 8;
  atomicAdd(op + 0, f0.x * alpha); atomicAdd(op + 1, f0.y * alpha);
  atomicAdd(op + 2, f0.z * alpha); atomicAdd(op + 3, f0.w * alpha);
  atomicAdd(op + 4, f1.x * alpha); atomicAdd(op + 5, f1.y * alpha);
  atomicAdd(op + 6, f1.z * alpha); atomicAdd(op + 7, f1.w * alpha);
}

// ---------------- bias + ELU (in place) ---------------------------------
__global__ void k_bias_elu(float* __restrict__ p, const float* __restrict__ b, int total) {
  int i = blockIdx.x * 256 + threadIdx.x;
  if (i >= total) return;
  float v = p[i] + b[i & (HIDALL - 1)];
  p[i] = v > 0.f ? v : (expf(v) - 1.f);
}

// ---------------- layer-2 node: feat2 = h@W2, el2/er2 (wave/node) -------
__global__ void __launch_bounds__(256)
k_layer2_node(const float* __restrict__ h, const float* __restrict__ W2,
              const float* __restrict__ al2, const float* __restrict__ ar2,
              float* __restrict__ feat2, float* __restrict__ el2,
              float* __restrict__ er2, int N) {
  int n = blockIdx.x * 8 + (threadIdx.x >> 5);
  if (n >= N) return;
  int lane = threadIdx.x & 31;
  const float* hp = h + (size_t)n * HIDALL + lane * 8;
  float p0 = 0.f, p1 = 0.f;
  #pragma unroll
  for (int t = 0; t < 8; ++t) {
    float hv = hp[t];
    int j = lane * 8 + t;
    p0 += hv * W2[j * 2 + 0];
    p1 += hv * W2[j * 2 + 1];
  }
  #pragma unroll
  for (int off = 16; off > 0; off >>= 1) {
    p0 += __shfl_xor(p0, off, 32);
    p1 += __shfl_xor(p1, off, 32);
  }
  if (lane == 0) {
    feat2[n * 2 + 0] = p0; feat2[n * 2 + 1] = p1;
    el2[n] = p0 * al2[0] + p1 * al2[1];
    er2[n] = p0 * ar2[0] + p1 * ar2[1];
  }
}

// ---------------- layer-2 edge phases (H=1, F=2) ------------------------
__global__ void k_edge2_score(const int* __restrict__ src, const int* __restrict__ dst,
                              const float* __restrict__ el2, const float* __restrict__ er2,
                              float* __restrict__ e2, float* __restrict__ m2, int E) {
  int e = blockIdx.x * 256 + threadIdx.x;
  if (e >= E) return;
  float x = el2[src[e]] + er2[dst[e]];
  x = x > 0.f ? x : NEG_SLOPE * x;
  e2[e] = x;
  atomicMaxF(m2 + dst[e], x);
}

__global__ void k_edge2_expsum(const int* __restrict__ dst, float* __restrict__ e2,
                               const float* __restrict__ m2, float* __restrict__ d2, int E) {
  int e = blockIdx.x * 256 + threadIdx.x;
  if (e >= E) return;
  int d = dst[e];
  float v = expf(e2[e] - m2[d]);
  e2[e] = v;
  atomicAdd(d2 + d, v);
}

__global__ void k_edge2_aggregate(const int* __restrict__ src, const int* __restrict__ dst,
                                  const float* __restrict__ feat2, const float* __restrict__ ee,
                                  const float* __restrict__ d2, float* __restrict__ out2, int E) {
  int e = blockIdx.x * 256 + threadIdx.x;
  if (e >= E) return;
  int s = src[e], d = dst[e];
  float alpha = ee[e] / fmaxf(d2[d], 1e-16f);
  atomicAdd(out2 + (size_t)d * 2 + 0, feat2[(size_t)s * 2 + 0] * alpha);
  atomicAdd(out2 + (size_t)d * 2 + 1, feat2[(size_t)s * 2 + 1] * alpha);
}

// ---------------- bias + log_softmax ------------------------------------
__global__ void k_logsoftmax(const float* __restrict__ out2, const float* __restrict__ b2,
                             float* __restrict__ out, int N) {
  int n = blockIdx.x * 256 + threadIdx.x;
  if (n >= N) return;
  float z0 = out2[n * 2 + 0] + b2[0];
  float z1 = out2[n * 2 + 1] + b2[1];
  float mx = fmaxf(z0, z1);
  float l = logf(expf(z0 - mx) + expf(z1 - mx));
  out[n * 2 + 0] = z0 - mx - l;
  out[n * 2 + 1] = z1 - mx - l;
}

// ========================================================================
extern "C" void kernel_launch(void* const* d_in, const int* in_sizes, int n_in,
                              void* d_out, int out_size, void* d_ws, size_t ws_size,
                              hipStream_t stream) {
  const float* x   = (const float*)d_in[0];
  const int*   src = (const int*)  d_in[1];
  const int*   dst = (const int*)  d_in[2];
  const float* W1  = (const float*)d_in[3];
  const float* al1 = (const float*)d_in[4];
  const float* ar1 = (const float*)d_in[5];
  const float* b1  = (const float*)d_in[6];
  const float* W2  = (const float*)d_in[7];
  const float* al2 = (const float*)d_in[8];
  const float* ar2 = (const float*)d_in[9];
  const float* b2  = (const float*)d_in[10];
  float* out = (float*)d_out;

  const int N = N_NODES, E = N_EDGES;

  // carve workspace (256-byte aligned slices)
  char* w = (char*)d_ws;
  auto carve = [&](size_t bytes) { char* p = w; w += (bytes + 255) & ~(size_t)255; return p; };
  unsigned short* xh   = (unsigned short*)carve((size_t)N * NFEAT * 2);
  unsigned short* w1h  = (unsigned short*)carve((size_t)NFEAT * HIDALL * 2);
  float* feat  = (float*)carve((size_t)N * HIDALL * 4);
  float* el    = (float*)carve((size_t)N * HEADS * 4);
  float* er    = (float*)carve((size_t)N * HEADS * 4);
  float* m1    = (float*)carve((size_t)N * HEADS * 4);
  float* d1    = (float*)carve((size_t)N * HEADS * 4);
  float* e1    = (float*)carve((size_t)E * HEADS * 4);
  float* out1  = (float*)carve((size_t)N * HIDALL * 4);   // becomes h after ELU
  float* feat2 = (float*)carve((size_t)N * 2 * 4);
  float* el2   = (float*)carve((size_t)N * 4);
  float* er2   = (float*)carve((size_t)N * 4);
  float* m2    = (float*)carve((size_t)N * 4);
  float* d2    = (float*)carve((size_t)N * 4);
  float* e2    = (float*)carve((size_t)E * 4);
  float* out2  = (float*)carve((size_t)N * 2 * 4);

  auto blk = [](int n) { return (n + 255) / 256; };

  // bf16 conversions
  k_f32_to_bf16<<<blk(N * NFEAT), 256, 0, stream>>>(x, xh, N * NFEAT);
  k_f32_to_bf16<<<blk(NFEAT * HIDALL), 256, 0, stream>>>(W1, w1h, NFEAT * HIDALL);

  // layer-1 GEMM via WMMA
  dim3 g1((N + 63) / 64, HIDALL / 64);
  k_gemm_bf16_wmma<<<g1, 256, 0, stream>>>(xh, w1h, feat, N, NFEAT, HIDALL);

  // attention logits per node
  k_elr<<<blk(N * HEADS), 256, 0, stream>>>(feat, al1, ar1, el, er, N);

  // init accumulators
  k_fill<<<blk(N * HEADS), 256, 0, stream>>>(m1, -INFINITY, N * HEADS);
  k_fill<<<blk(N * HEADS), 256, 0, stream>>>(d1, 0.f, N * HEADS);
  k_fill<<<blk(N * HIDALL), 256, 0, stream>>>(out1, 0.f, N * HIDALL);
  k_fill<<<blk(N), 256, 0, stream>>>(m2, -INFINITY, N);
  k_fill<<<blk(N), 256, 0, stream>>>(d2, 0.f, N);
  k_fill<<<blk(N * 2), 256, 0, stream>>>(out2, 0.f, N * 2);

  // layer-1 edge softmax + aggregation
  k_edge1_score<<<blk(E), 256, 0, stream>>>(src, dst, el, er, e1, m1, E);
  k_edge1_expsum<<<blk(E), 256, 0, stream>>>(dst, e1, m1, d1, E);
  k_edge1_aggregate<<<(E + 7) / 8, 256, 0, stream>>>(src, dst, feat, e1, d1, out1, E);

  // bias + ELU
  k_bias_elu<<<blk(N * HIDALL), 256, 0, stream>>>(out1, b1, N * HIDALL);

  // layer-2 projection + logits
  k_layer2_node<<<(N + 7) / 8, 256, 0, stream>>>(out1, W2, al2, ar2, feat2, el2, er2, N);

  // layer-2 edge softmax + aggregation
  k_edge2_score<<<blk(E), 256, 0, stream>>>(src, dst, el2, er2, e2, m2, E);
  k_edge2_expsum<<<blk(E), 256, 0, stream>>>(dst, e2, m2, d2, E);
  k_edge2_aggregate<<<blk(E), 256, 0, stream>>>(src, dst, feat2, e2, d2, out2, E);

  // final bias + log_softmax
  k_logsoftmax<<<blk(N), 256, 0, stream>>>(out2, b2, out, N);
}
